// QCNNGen375_65481071406866
// MI455X (gfx1250) — matrix operand order repaired and checked by reading, not compiled, not run on hardware
//
#include <hip/hip_runtime.h>
#include <hip/hip_bf16.h>
#include <math.h>

typedef __attribute__((ext_vector_type(16))) _Float16 v16h;
typedef __attribute__((ext_vector_type(8)))  float    v8f;

#define NQ     8
#define DIM    256      // 2^NQ
#define NLAYER 2
#define NSTAGE 8        // 8 stages x 32 state-rows = 256 states
#define ROWS_PER_STAGE 32

// ---------------------------------------------------------------------------
// CDNA5 async global->LDS copy (ASYNCcnt path), per-lane 16B transfer.
// LDS byte address = low 32 bits of the generic pointer (ISA 10.2: flat->LDS
// truncates to addr[31:0]).
// ---------------------------------------------------------------------------
__device__ __forceinline__ void async_copy_b128(const void* gptr, void* lptr) {
    unsigned loff = (unsigned)(size_t)lptr;
    asm volatile("global_load_async_to_lds_b128 %0, %1, off"
                 :: "v"(loff), "v"(gptr)
                 : "memory");
}
__device__ __forceinline__ void wait_async0() {
    asm volatile("s_wait_asynccnt 0x0" ::: "memory");
}

// ---------------------------------------------------------------------------
// Kernel 1: compose the full circuit unitary, one basis-state simulation per
// block.  Column k of U (i.e. circuit(e_k)) is written as f16 into row-major
// (s, k) arrays Urh/Uih so kernel 2's B-fragments are contiguous 32B loads.
// ---------------------------------------------------------------------------
__global__ __launch_bounds__(128) void qcnn_build_unitary(
    const float* __restrict__ w,          // (NLAYER, NQ) angles
    _Float16* __restrict__ Urh,           // (DIM, DIM) row-major (s, k)
    _Float16* __restrict__ Uih)           // (DIM, DIM)
{
    __shared__ float sr[DIM];
    __shared__ float si[DIM];
    const int tid = threadIdx.x;
    const int col = blockIdx.x;           // basis state index k

    for (int i = tid; i < DIM; i += 128) { sr[i] = 0.f; si[i] = 0.f; }
    __syncthreads();
    if (tid == 0) sr[col] = 1.f;

    for (int layer = 0; layer < NLAYER; ++layer) {
        for (int rep = 0; rep < 2; ++rep) {
            // RX sweep over all qubits (same weights both sweeps)
            for (int q = 0; q < NQ; ++q) {
                const float th  = w[layer * NQ + q] * 0.5f;
                const float cth = cosf(th);
                const float sth = sinf(th);
                const int   p   = 7 - q;              // bit position of qubit q
                __syncthreads();
                // each of the 128 threads owns one disjoint amplitude pair
                const unsigned bit  = 1u << p;
                const unsigned low  = (unsigned)tid & (bit - 1u);
                const unsigned high = ((unsigned)tid >> p) << (p + 1);
                const unsigned s0 = high | low;
                const unsigned s1 = s0 | bit;
                const float a0r = sr[s0], a0i = si[s0];
                const float a1r = sr[s1], a1i = si[s1];
                // RX = [[c, -i s], [-i s, c]]
                sr[s0] = cth * a0r + sth * a1i;
                si[s0] = cth * a0i - sth * a1r;
                sr[s1] = cth * a1r + sth * a0i;
                si[s1] = cth * a1i - sth * a0r;
            }
            if (rep == 0) {
                // CNOT chain: control c (bit 7-c) -> target c+1 (bit 6-c)
                for (int c = 0; c < NQ - 1; ++c) {
                    const int pc = 7 - c;
                    const int pt = pc - 1;
                    __syncthreads();
                    if (tid < 64) {
                        const unsigned low  = (unsigned)tid & ((1u << pt) - 1u);
                        const unsigned high = ((unsigned)tid >> pt) << (pt + 2);
                        const unsigned s0 = high | (1u << pc) | low;   // ctrl=1, tgt=0
                        const unsigned s1 = s0 | (1u << pt);           // ctrl=1, tgt=1
                        float tr = sr[s0]; sr[s0] = sr[s1]; sr[s1] = tr;
                        float ti = si[s0]; si[s0] = si[s1]; si[s1] = ti;
                    }
                }
            }
        }
    }
    __syncthreads();
    for (int i = tid; i < DIM; i += 128) {
        Urh[(size_t)i * DIM + col] = (_Float16)sr[i];
        Uih[(size_t)i * DIM + col] = (_Float16)si[i];
    }
}

// ---------------------------------------------------------------------------
// Kernel 2: fused  Yr = X Ur^T, Yi = X Ui^T  via v_wmma_f32_16x16x32_f16 with
// U double-buffer-staged into LDS via global_load_async_to_lds_b128, then
// signed |Y|^2 reductions -> Z expectations -> MLP -> sigmoid.
// One wave handles 16 batch rows. 8 waves / block, 64 blocks -> 8192 rows.
// ---------------------------------------------------------------------------
__global__ __launch_bounds__(256) void qcnn_wmma_fused(
    const float* __restrict__ x,          // (8192, 256)
    const _Float16* __restrict__ Urh,     // (256, 256) (s, k)
    const _Float16* __restrict__ Uih,
    const float* __restrict__ W1,         // (16, 8)
    const float* __restrict__ b1,         // (16)
    const float* __restrict__ W2,         // (1, 16)
    const float* __restrict__ b2,         // (1)
    float* __restrict__ out)              // (8192)
{
    // 2 buffers x {re,im} x 32 state-rows x 256 k of f16 = exactly 64 KB.
    __shared__ _Float16 ubuf[2][2][ROWS_PER_STAGE][DIM];

    const int tid  = threadIdx.x;
    const int warp = tid >> 5;
    const int lane = tid & 31;
    const int half = lane >> 4;           // 0: lanes 0-15, 1: lanes 16-31
    const int n    = lane & 15;           // column-within-tile / row M index
    const int row0 = (blockIdx.x * 8 + warp) * 16;

    const float* xrow = x + (size_t)(row0 + n) * DIM;

    // -------- issue async fill of stage 0 --------
    // stage data = 2048 x 16B chunks; thread tid copies chunks tid + 256*j.
    #pragma unroll
    for (int j = 0; j < 8; ++j) {
        const int linear = tid + 256 * j;         // 0..2047
        const int ri  = linear >> 10;             // uniform per j
        const int idx = linear & 1023;            // 16B-chunk index within block
        const _Float16* g = (ri ? Uih : Urh) + (size_t)idx * 8;
        _Float16* l = &ubuf[0][0][0][0] + (size_t)ri * (ROWS_PER_STAGE * DIM)
                                        + (size_t)idx * 8;
        async_copy_b128(g, l);
    }

    // -------- A fragments + fused row sum-of-squares --------
    // A k order used here (same as B): lanes<16 hold K k0+0..15,
    //                        lanes>=16 hold K k0+16..31.
    // Lane l and lane l+16 together touch every element of row l exactly once,
    // so ||x_row||^2 falls out of the same loads (shfl_xor 16 to combine).
    float nsq = 0.f;
    v16h afrag[8];
    #pragma unroll
    for (int c = 0; c < 8; ++c) {
        const int ka = 32 * c + half * 16;
        const int kb = 32 * c + half * 16 + 8;
        #pragma unroll
        for (int j = 0; j < 8; ++j) {
            const float va = xrow[ka + j];
            const float vb = xrow[kb + j];
            nsq = fmaf(va, va, nsq);
            nsq = fmaf(vb, vb, nsq);
            afrag[c][j]     = (_Float16)va;
            afrag[c][8 + j] = (_Float16)vb;
        }
    }
    nsq += __shfl_xor(nsq, 16);

    // signed probability accumulators: 8 local rows (C vgpr index) x 8 qubits
    float zacc[64];
    #pragma unroll
    for (int i = 0; i < 64; ++i) zacc[i] = 0.f;

    // stage 0 ready: own loads done, then block-wide visibility
    wait_async0();
    __syncthreads();

    for (int st = 0; st < NSTAGE; ++st) {
        const int b = st & 1;
        // issue next stage into the other buffer (dead since stage st-1's
        // readers passed the barrier that ended the previous iteration)
        if (st + 1 < NSTAGE) {
            const size_t gbase = (size_t)ROWS_PER_STAGE * (st + 1) * DIM;
            _Float16* lbase = &ubuf[b ^ 1][0][0][0];
            #pragma unroll
            for (int j = 0; j < 8; ++j) {
                const int linear = tid + 256 * j;
                const int ri  = linear >> 10;
                const int idx = linear & 1023;
                const _Float16* g = (ri ? Uih : Urh) + gbase + (size_t)idx * 8;
                _Float16* l = lbase + (size_t)ri * (ROWS_PER_STAGE * DIM)
                                    + (size_t)idx * 8;
                async_copy_b128(g, l);
            }
        }

        // compute the 2 N-tiles of this stage from LDS
        #pragma unroll
        for (int tl = 0; tl < 2; ++tl) {
            const int s  = (st * 2 + tl) * 16 + n;   // global state index
            const int lr = tl * 16 + n;              // local row in stage
            const _Float16* brp = &ubuf[b][0][lr][half * 16];
            const _Float16* bip = &ubuf[b][1][lr][half * 16];
            v8f cr = {};
            v8f ci = {};
            #pragma unroll
            for (int c = 0; c < 8; ++c) {
                const v16h br = *(const v16h*)(brp + 32 * c);
                const v16h bi = *(const v16h*)(bip + 32 * c);
                cr = __builtin_amdgcn_wmma_f32_16x16x32_f16(
                        false, afrag[c], false, br, (short)0, cr, false, false);
                ci = __builtin_amdgcn_wmma_f32_16x16x32_f16(
                        false, afrag[c], false, bi, (short)0, ci, false, false);
            }
            // epilogue for this N-tile: p = |amp|^2, accumulate with Z signs
            float sgn[NQ];
            #pragma unroll
            for (int q = 0; q < NQ; ++q)
                sgn[q] = ((s >> (7 - q)) & 1) ? -1.f : 1.f;
            #pragma unroll
            for (int v = 0; v < 8; ++v) {
                const float p = cr[v] * cr[v] + ci[v] * ci[v];
                #pragma unroll
                for (int q = 0; q < NQ; ++q)
                    zacc[v * 8 + q] = fmaf(sgn[q], p, zacc[v * 8 + q]);
            }
        }

        if (st + 1 < NSTAGE) wait_async0();   // next buffer landed (own part)
        __syncthreads();                       // block-wide: landed + consumed
    }

    // reduce across the 16 lanes of each half (each half owns 8 batch rows)
    #pragma unroll
    for (int off = 1; off < 16; off <<= 1) {
        #pragma unroll
        for (int i = 0; i < 64; ++i)
            zacc[i] += __shfl_xor(zacc[i], off);
    }

    // stage Z sums and norms through LDS (reuse ubuf: safe after the final
    // barrier above) so lanes 0-15 can run the MLP one batch-row per lane
    float* zs = (float*)&ubuf[0][0][0][0];      // [8 warps][16 rows][8 q]
    float* ns = zs + 8 * 16 * NQ;               // [8 warps][16 rows]
    if (n == 0) {                  // lane 0 -> rows 0..7, lane 16 -> rows 8..15
        #pragma unroll
        for (int m = 0; m < 8; ++m)
            #pragma unroll
            for (int q = 0; q < NQ; ++q)
                zs[((warp * 16) + half * 8 + m) * NQ + q] = zacc[m * 8 + q];
    }
    if (half == 0) ns[warp * 16 + n] = nsq;
    __syncthreads();

    if (half == 0) {
        const float inv = 1.f / ns[warp * 16 + n];
        float z[NQ];
        #pragma unroll
        for (int q = 0; q < NQ; ++q) z[q] = zs[(warp * 16 + n) * NQ + q] * inv;
        float h[16];
        #pragma unroll
        for (int j = 0; j < 16; ++j) {
            float acc = b1[j];
            #pragma unroll
            for (int q = 0; q < NQ; ++q) acc = fmaf(W1[j * NQ + q], z[q], acc);
            h[j] = tanhf(acc);
        }
        float acc = b2[0];
        #pragma unroll
        for (int j = 0; j < 16; ++j) acc = fmaf(W2[j], h[j], acc);
        out[row0 + n] = 1.f / (1.f + expf(-acc));
    }
}

// ---------------------------------------------------------------------------
extern "C" void kernel_launch(void* const* d_in, const int* in_sizes, int n_in,
                              void* d_out, int out_size, void* d_ws, size_t ws_size,
                              hipStream_t stream) {
    const float* x          = (const float*)d_in[0];   // (8192, 256)
    const float* var_params = (const float*)d_in[1];   // (2, 8)
    const float* W1         = (const float*)d_in[2];   // (16, 8)
    const float* b1         = (const float*)d_in[3];   // (16)
    const float* W2         = (const float*)d_in[4];   // (1, 16)
    const float* b2         = (const float*)d_in[5];   // (1)
    float*       out        = (float*)d_out;           // (8192)

    _Float16* Urh = (_Float16*)d_ws;                   // 256*256 f16 = 128 KB
    _Float16* Uih = Urh + (size_t)DIM * DIM;           // 128 KB more

    qcnn_build_unitary<<<DIM, 128, 0, stream>>>(var_params, Urh, Uih);

    const int B = in_sizes[0] / DIM;                   // 8192
    const int blocks = B / (8 * 16);                   // 8 waves x 16 rows per block
    qcnn_wmma_fused<<<blocks, 256, 0, stream>>>(x, Urh, Uih, W1, b1, W2, b2, out);
}
